// SAE_5995774345917
// MI455X (gfx1250) — matrix operand order repaired and checked by reading, not compiled, not run on hardware
//
#include <hip/hip_runtime.h>

// ---------------------------------------------------------------------------
// Types for WMMA (CDNA5 / gfx1250, wave32)
// ---------------------------------------------------------------------------
typedef __attribute__((ext_vector_type(4)))  unsigned short u16x4;
typedef __attribute__((ext_vector_type(8)))  unsigned short u16x8;
typedef __attribute__((ext_vector_type(16))) unsigned short u16x16;
typedef __attribute__((ext_vector_type(4)))  __bf16         bf16x4;
typedef __attribute__((ext_vector_type(16))) __bf16         bf16x16;
typedef __attribute__((ext_vector_type(8)))  float          f32x8;

// Native f32->bf16 conversion: clang lowers to v_cvt_pk_bf16_f32 on gfx1250
// (1 VALU per 2 elements vs ~7 for the manual RNE bit sequence).
__device__ __forceinline__ void cvt_store4(unsigned short* dst, float4 f) {
  bf16x4 p = { (__bf16)f.x, (__bf16)f.y, (__bf16)f.z, (__bf16)f.w };
  *(u16x4*)dst = __builtin_bit_cast(u16x4, p);
}

// ---------------------------------------------------------------------------
// NT GEMM:  C[M,N] = A[M,K] * W[N,K]^T (+bias[N]) (+ReLU)
// A, W row-major f32 (K contiguous for both). bf16 WMMA, f32 accumulate.
// Block tile 128x128, 8 waves, wave tile 32x64 (2x4 frags of 16x16x32).
// M % 128 == 0, N % 128 == 0, K % 32 == 0 (true for all calls here).
// nt_out=1 streams C with TH=NT so huge outputs don't evict L2-resident
// weights (192 MB global L2; enc/dec weights are reused by 32 M-blocks).
// ---------------------------------------------------------------------------
#define TM   128
#define TN   128
#define TK   32
#define LDSW 40   // shorts per LDS row: 32 data + 8 pad (80B: 16B-aligned, bank-spread)

__global__ void __launch_bounds__(256)
sae_gemm_bf16_nt(const float* __restrict__ A, const float* __restrict__ W,
                 const float* __restrict__ bias, float* __restrict__ C,
                 int M, int N, int K, int relu, int nt_out)
{
  __shared__ unsigned short As[2][TM * LDSW];
  __shared__ unsigned short Bs[2][TN * LDSW];

  const int tid  = threadIdx.x;
  const int lane = tid & 31;
  const int wave = tid >> 5;
  const int half = lane >> 4;   // lane group 0..15 vs 16..31 (ISA 16-bit frag layout)
  const int l16  = lane & 15;

  const int bm = blockIdx.y * TM;
  const int bn = blockIdx.x * TN;
  const int wm = (wave >> 1) * 32;   // waves arranged 4(M) x 2(N)
  const int wn = (wave & 1) * 64;

  // global staging: each thread loads a 16-float half-row of A and of W
  const int lr = tid >> 1;           // tile row 0..127
  const int lc = (tid & 1) * 16;     // col offset (floats)

  const float* Ag = A + (size_t)(bm + lr) * K + lc;
  const float* Wg = W + (size_t)(bn + lr) * K + lc;

  f32x8 acc[2][4];
#pragma unroll
  for (int i = 0; i < 2; ++i)
#pragma unroll
    for (int j = 0; j < 4; ++j)
      acc[i][j] = (f32x8)0.0f;

  const int nk = K / TK;
  float4 ra[4], rb[4];

  // ---- stage tile 0
#pragma unroll
  for (int i = 0; i < 4; ++i) {
    ra[i] = ((const float4*)Ag)[i];
    rb[i] = ((const float4*)Wg)[i];
  }
#pragma unroll
  for (int i = 0; i < 4; ++i) {
    cvt_store4(&As[0][lr * LDSW + lc + i * 4], ra[i]);
    cvt_store4(&Bs[0][lr * LDSW + lc + i * 4], rb[i]);
  }
  __syncthreads();

  int buf = 0;
  for (int kt = 0; kt < nk; ++kt) {
    // global loads for next tile (overlap with compute)
    if (kt + 1 < nk) {
      const float4* ap = (const float4*)(Ag + (size_t)(kt + 1) * TK);
      const float4* wp = (const float4*)(Wg + (size_t)(kt + 1) * TK);
#pragma unroll
      for (int i = 0; i < 4; ++i) { ra[i] = ap[i]; rb[i] = wp[i]; }
    }
    // L2 prefetch two tiles ahead (gfx1250 global_prefetch_b8)
    if (kt + 2 < nk) {
      __builtin_prefetch(Ag + (size_t)(kt + 2) * TK, 0, 3);
      __builtin_prefetch(Wg + (size_t)(kt + 2) * TK, 0, 3);
    }

    // ---- fragment loads per ISA 16-bit layouts, then 8 WMMAs
    bf16x16 af[2], bv[4];
#pragma unroll
    for (int i = 0; i < 2; ++i) {
      // A 16x32: lane = M row; half selects K[8h..8h+7] then K[16+8h..]
      const unsigned short* p = &As[buf][(wm + i * 16 + l16) * LDSW];
      u16x8 lo = *(const u16x8*)(p + 8 * half);
      u16x8 hi = *(const u16x8*)(p + 16 + 8 * half);
      af[i] = __builtin_bit_cast(bf16x16,
              __builtin_shufflevector(lo, hi, 0,1,2,3,4,5,6,7,8,9,10,11,12,13,14,15));
    }
#pragma unroll
    for (int j = 0; j < 4; ++j) {
      // B 32x16: lane = N col; half selects K[8h..8h+7] then K[16+8h..]
      const unsigned short* p = &Bs[buf][(wn + j * 16 + l16) * LDSW];
      u16x8 lo = *(const u16x8*)(p + 8 * half);
      u16x8 hi = *(const u16x8*)(p + 16 + 8 * half);
      bv[j] = __builtin_bit_cast(bf16x16,
              __builtin_shufflevector(lo, hi, 0,1,2,3,4,5,6,7,8,9,10,11,12,13,14,15));
    }
#pragma unroll
    for (int i = 0; i < 2; ++i)
#pragma unroll
      for (int j = 0; j < 4; ++j)
        acc[i][j] = __builtin_amdgcn_wmma_f32_16x16x32_bf16(
            false, af[i], false, bv[j], (short)0, acc[i][j], false, false);

    // ---- stage next tile into the other buffer
    if (kt + 1 < nk) {
      const int nb = buf ^ 1;
#pragma unroll
      for (int i = 0; i < 4; ++i) {
        cvt_store4(&As[nb][lr * LDSW + lc + i * 4], ra[i]);
        cvt_store4(&Bs[nb][lr * LDSW + lc + i * 4], rb[i]);
      }
    }
    __syncthreads();
    buf ^= 1;
  }

  // ---- epilogue: bias + optional ReLU; 16x16 f32 C/D layout (8 VGPRs)
#pragma unroll
  for (int j = 0; j < 4; ++j) {
    const int col = bn + wn + j * 16 + l16;
    const float bvv = bias ? bias[col] : 0.0f;
#pragma unroll
    for (int i = 0; i < 2; ++i) {
      const int row0 = bm + wm + i * 16 + half * 8;
#pragma unroll
      for (int r = 0; r < 8; ++r) {
        float v = acc[i][j][r] + bvv;
        if (relu) v = fmaxf(v, 0.0f);
        float* dst = &C[(size_t)(row0 + r) * N + col];
        if (nt_out) __builtin_nontemporal_store(v, dst);
        else        *dst = v;
      }
    }
  }
}

// ---------------------------------------------------------------------------
// Exact in-place per-row top-k (H = 16384). ReLU output is >= 0, so f32
// ordering == u32 ordering of the bit patterns. 4-pass radix select with a
// 256-bin LDS histogram gives the exact k-th largest pattern + tie budget.
// ---------------------------------------------------------------------------
#define TOPK_H 16384
#define TOPK_T 256
#define TOPK_E (TOPK_H / TOPK_T)   // 64 elements per thread, held in VGPRs

__global__ void __launch_bounds__(TOPK_T)
sae_topk_inplace(float* __restrict__ latent, const int* __restrict__ kptr)
{
  __shared__ unsigned hist[256];
  __shared__ unsigned s_prefix, s_need;
  __shared__ int s_ties;

  const int tid = threadIdx.x;
  float* row = latent + (size_t)blockIdx.x * TOPK_H;

  unsigned uv[TOPK_E];
#pragma unroll
  for (int c = 0; c < TOPK_E; ++c)
    uv[c] = __float_as_uint(row[tid + c * TOPK_T]);

  unsigned prefix = 0;
  unsigned need = (unsigned)(*kptr);

  for (int shift = 24; shift >= 0; shift -= 8) {
    hist[tid] = 0;
    __syncthreads();
    const unsigned maskHi = (shift == 24) ? 0u : (0xFFFFFFFFu << (shift + 8));
#pragma unroll
    for (int c = 0; c < TOPK_E; ++c) {
      const unsigned u = uv[c];
      if ((u & maskHi) == prefix)
        atomicAdd(&hist[(u >> shift) & 0xFFu], 1u);
    }
    __syncthreads();
    if (tid == 0) {
      unsigned cum = 0;
      for (int b = 255; b >= 0; --b) {
        const unsigned c = hist[b];
        if (cum + c >= need) {
          s_prefix = prefix | ((unsigned)b << shift);
          s_need = need - cum;
          break;
        }
        cum += c;
      }
    }
    __syncthreads();
    prefix = s_prefix;
    need = s_need;
    __syncthreads();
  }

  // prefix == bit pattern of k-th largest; need == #ties at threshold to keep
  if (tid == 0) s_ties = (int)need;
  __syncthreads();
#pragma unroll
  for (int c = 0; c < TOPK_E; ++c) {
    const unsigned u = uv[c];
    if (u > prefix) continue;                    // strictly above threshold: keep
    if (u == prefix) {
      const int old = atomicSub(&s_ties, 1);
      if (old > 0) continue;                     // kept tie
    }
    row[tid + c * TOPK_T] = 0.0f;                // dropped: zero in place
  }
}

// ---------------------------------------------------------------------------
// Host-side orchestration
// ---------------------------------------------------------------------------
static void launch_gemm(const float* A, const float* W, const float* bias,
                        float* C, int M, int N, int K, int relu, int nt_out,
                        hipStream_t stream)
{
  dim3 grid((unsigned)(N / TN), (unsigned)(M / TM));
  dim3 blk(256);
  hipLaunchKernelGGL(sae_gemm_bf16_nt, grid, blk, 0, stream,
                     A, W, bias, C, M, N, K, relu, nt_out);
}

extern "C" void kernel_launch(void* const* d_in, const int* in_sizes, int n_in,
                              void* d_out, int out_size, void* d_ws, size_t ws_size,
                              hipStream_t stream)
{
  (void)in_sizes; (void)n_in; (void)out_size; (void)d_ws; (void)ws_size;

  const float* img        = (const float*)d_in[0];
  const float* txt        = (const float*)d_in[1];
  const float* w_img_proj = (const float*)d_in[2];
  const float* w_txt_proj = (const float*)d_in[3];
  const float* w_img_enc  = (const float*)d_in[4];
  const float* b_img_enc  = (const float*)d_in[5];
  const float* w_txt_enc  = (const float*)d_in[6];
  const float* b_txt_enc  = (const float*)d_in[7];
  const float* w_img_dec  = (const float*)d_in[8];
  const float* b_img_dec  = (const float*)d_in[9];
  const float* w_txt_dec  = (const float*)d_in[10];
  const float* b_txt_dec  = (const float*)d_in[11];
  const int*   kptr       = (const int*)d_in[12];

  const int B = 4096, P = 1024, H = 16384;
  const int D_IMG = 1024, D_TXT = 768;

  float* out        = (float*)d_out;
  float* o_img_proj = out;
  float* o_img_lat  = o_img_proj + (size_t)B * P;
  float* o_img_rec  = o_img_lat  + (size_t)B * H;
  float* o_txt_proj = o_img_rec  + (size_t)B * P;
  float* o_txt_lat  = o_txt_proj + (size_t)B * P;
  float* o_txt_rec  = o_txt_lat  + (size_t)B * H;

  // ---- image branch
  // proj output (16 MB) is immediately re-read by the encoder -> temporal store
  launch_gemm(img,        w_img_proj, nullptr,   o_img_proj, B, P, D_IMG, 0, 0, stream);
  // latent (256 MB > L2) -> NT store, keep 64 MB encoder weights resident
  launch_gemm(o_img_proj, w_img_enc,  b_img_enc, o_img_lat,  B, H, P,     1, 1, stream);
  hipLaunchKernelGGL(sae_topk_inplace, dim3((unsigned)B), dim3(TOPK_T), 0, stream,
                     o_img_lat, kptr);
  // recon is final output -> NT store
  launch_gemm(o_img_lat,  w_img_dec,  b_img_dec, o_img_rec,  B, P, H,     0, 1, stream);

  // ---- text branch
  launch_gemm(txt,        w_txt_proj, nullptr,   o_txt_proj, B, P, D_TXT, 0, 0, stream);
  launch_gemm(o_txt_proj, w_txt_enc,  b_txt_enc, o_txt_lat,  B, H, P,     1, 1, stream);
  hipLaunchKernelGGL(sae_topk_inplace, dim3((unsigned)B), dim3(TOPK_T), 0, stream,
                     o_txt_lat, kptr);
  launch_gemm(o_txt_lat,  w_txt_dec,  b_txt_dec, o_txt_rec,  B, P, H,     0, 1, stream);
}
